// PACNet_88845693485066
// MI455X (gfx1250) — matrix operand order repaired and checked by reading, not compiled, not run on hardware
//
#include <hip/hip_runtime.h>
#include <hip/hip_bf16.h>
#include <math.h>

typedef __attribute__((ext_vector_type(16))) _Float16 v16h;
typedef __attribute__((ext_vector_type(8)))  float    v8f;

#define NEG_SLOPE 0.01f

// ---------------------------------------------------------------------------
// Kernel 1: 1x1 conv 384->16 over 4096 positions as WMMA GEMM.
//   Y(16 x 4096) = W(16 x 384) * X(384 x 4096), +bias, leaky ReLU.
// One wave per 16x16 output tile (M=16 out-channels, N=16 positions),
// K marched in steps of 32 with v_wmma_f32_16x16x32_f16 (f32 accumulate).
// Fragment layouts per CDNA5 ISA 7.12.2 (wave32):
//   A 16x32 f16: lane -> M=lane%16; half j (v=j/2,p=j&1):
//       K = (v<4 ? 2v+p : 16+2(v-4)+p) + 8*(lane/16)
//   B 32x16 f16: lane -> N=lane%16; K = (v<4 ? 2v+p : 16+2(v-4)+p) + 8*(lane/16)
//   C 16x16 f32: vgpr r -> M = r + 8*(lane/16), N = lane%16
// ---------------------------------------------------------------------------
__global__ void lin1x1_wmma(const float* __restrict__ xin,   // (4,384,32,32)
                            const float* __restrict__ lw,    // (16,384)
                            const float* __restrict__ lb,    // (16)
                            float* __restrict__ yout)        // (4,16,32,32)
{
    const int lane = threadIdx.x & 31;
    const int wave = threadIdx.x >> 5;
    const int tile = blockIdx.x * 4 + wave;   // 256 tiles = 4 batches * 64
    const int b    = tile >> 6;
    const int pos0 = (tile & 63) << 4;
    const int g    = lane >> 4;
    const int m    = lane & 15;

    v8f c = {};
    for (int k0 = 0; k0 < 384; k0 += 32) {
        v16h afrag, bfrag;
        #pragma unroll
        for (int j = 0; j < 16; ++j) {
            const int v = j >> 1, p = j & 1;
            const int ka = ((v < 4) ? (2 * v + p) : (16 + 2 * (v - 4) + p)) + 8 * g;
            afrag[j] = (_Float16)lw[m * 384 + k0 + ka];
            const int kb = ((v < 4) ? (2 * v + p) : (16 + 2 * (v - 4) + p)) + 8 * g;
            bfrag[j] = (_Float16)xin[(b * 384 + k0 + kb) * 1024 + pos0 + m];
        }
        c = __builtin_amdgcn_wmma_f32_16x16x32_f16(false, afrag, false, bfrag,
                                                   (short)0, c, false, false);
    }
    #pragma unroll
    for (int r = 0; r < 8; ++r) {
        const int mo = r + 8 * g;
        float v = c[r] + lb[mo];
        v = (v >= 0.f) ? v : NEG_SLOPE * v;
        yout[(b * 16 + mo) * 1024 + pos0 + m] = v;
    }
}

// ---------------------------------------------------------------------------
// PAC stage: fused (up2-nearest of x) + (bilinear resize of guide) +
// per-pixel Gaussian affinity + 3x3 kernel-modulated conv + activation.
// One thread per output pixel (all CO channels).
// ---------------------------------------------------------------------------
struct Axis { int i0, i1; float w; bool ok; };

__device__ inline Axis mkaxis(int q, int lim, float scale) {
    Axis a;
    a.ok = (q >= 0) && (q < lim);
    float c = fmaxf(((float)q + 0.5f) * scale - 0.5f, 0.f);
    int i0 = (int)c;
    a.i0 = i0;
    a.i1 = min(i0 + 1, 511);
    a.w  = c - (float)i0;
    return a;
}

__device__ inline float bilin(const float* __restrict__ gp,
                              const Axis& r, const Axis& cc) {
    float v00 = gp[r.i0 * 512 + cc.i0];
    float v01 = gp[r.i0 * 512 + cc.i1];
    float v10 = gp[r.i1 * 512 + cc.i0];
    float v11 = gp[r.i1 * 512 + cc.i1];
    float t0 = v00 + (v01 - v00) * cc.w;
    float t1 = v10 + (v11 - v10) * cc.w;
    return t0 + (t1 - t0) * r.w;
}

template<int CI, int CO, bool LAST>
__global__ void pac_stage(const float* __restrict__ xin,   // (4,CI,Hin,Hin)
                          int Hin,
                          const float* __restrict__ guide, // (4,3,512,512)
                          const float* __restrict__ wts,   // (CO,CI,3,3)
                          float* __restrict__ out)         // (4,CO,2Hin,2Hin)
{
    const int Hout = Hin * 2;
    const int HW   = Hout * Hout;

    __shared__ float sw[CO * CI * 9];
    for (int i = threadIdx.x; i < CO * CI * 9; i += blockDim.x) sw[i] = wts[i];
    __syncthreads();

    const int gid = blockIdx.x * blockDim.x + threadIdx.x;
    if (gid >= 4 * HW) return;
    const int b = gid / HW;
    const int p = gid - b * HW;
    const int h = p / Hout;
    const int w = p - h * Hout;

    const float scale = 512.0f / (float)Hout;

    Axis rows[3], cols[3];
    #pragma unroll
    for (int d = 0; d < 3; ++d) {
        rows[d] = mkaxis(h + d - 1, Hout, scale);
        cols[d] = mkaxis(w + d - 1, Hout, scale);
    }

    const float* gb = guide + (size_t)b * 3 * 512 * 512;
    float gc[3];
    #pragma unroll
    for (int cch = 0; cch < 3; ++cch)
        gc[cch] = bilin(gb + cch * 512 * 512, rows[1], cols[1]);

    float acc[CO];
    #pragma unroll
    for (int o = 0; o < CO; ++o) acc[o] = 0.f;

    const int HWin = Hin * Hin;

    #pragma unroll
    for (int dy = 0; dy < 3; ++dy) {
        #pragma unroll
        for (int dx = 0; dx < 3; ++dx) {
            // zero-padded tap: x_unf == 0 there, contributes nothing
            if (!(rows[dy].ok && cols[dx].ok)) continue;
            const int t = dy * 3 + dx;
            float d2 = 0.f;
            #pragma unroll
            for (int cch = 0; cch < 3; ++cch) {
                float gt = bilin(gb + cch * 512 * 512, rows[dy], cols[dx]);
                float dd = gt - gc[cch];
                d2 += dd * dd;
            }
            const float kv = expf(-0.5f * d2);
            const int xr = (h + dy - 1) >> 1;   // up2-nearest inverse
            const int xc = (w + dx - 1) >> 1;
            const float* xp = xin + (size_t)(b * CI) * HWin + xr * Hin + xc;
            #pragma unroll
            for (int i = 0; i < CI; ++i) {
                const float xk = xp[(size_t)i * HWin] * kv;
                #pragma unroll
                for (int o = 0; o < CO; ++o)
                    acc[o] += sw[(o * CI + i) * 9 + t] * xk;
            }
        }
    }

    if (LAST) {
        out[(size_t)b * HW + p] = 1.f / (1.f + expf(-acc[0]));
    } else {
        #pragma unroll
        for (int o = 0; o < CO; ++o) {
            float v = acc[o];
            out[((size_t)b * CO + o) * HW + p] = (v >= 0.f) ? v : NEG_SLOPE * v;
        }
    }
}

// ---------------------------------------------------------------------------
extern "C" void kernel_launch(void* const* d_in, const int* in_sizes, int n_in,
                              void* d_out, int out_size, void* d_ws, size_t ws_size,
                              hipStream_t stream) {
    const float* x     = (const float*)d_in[0]; // (4,384,32,32)
    const float* guide = (const float*)d_in[1]; // (4,3,512,512)
    const float* lin_w = (const float*)d_in[2]; // (16,384,1,1)
    const float* lin_b = (const float*)d_in[3]; // (16)
    const float* w0    = (const float*)d_in[4]; // (4,16,3,3)
    const float* w1    = (const float*)d_in[5]; // (4,4,3,3)
    const float* w2    = (const float*)d_in[6]; // (4,4,3,3)
    const float* w3    = (const float*)d_in[7]; // (1,4,3,3)
    float* outp = (float*)d_out;                // (4,1,512,512)

    char* ws = (char*)d_ws;
    float* buf_lin = (float*)(ws);                       // (4,16,32,32)   256 KB
    float* buf_s0  = (float*)(ws + 262144);              // (4,4,64,64)    256 KB
    float* buf_s1  = (float*)(ws + 524288);              // (4,4,128,128)    1 MB
    float* buf_s2  = (float*)(ws + 1572864);             // (4,4,256,256)    4 MB

    // 1x1 conv via WMMA: 256 tiles, 4 waves/block
    lin1x1_wmma<<<64, 128, 0, stream>>>(x, lin_w, lin_b, buf_lin);

    // Stage 0: 32 -> 64, CI=16 CO=4, leaky
    pac_stage<16, 4, false><<<(4 * 64 * 64 + 255) / 256, 256, 0, stream>>>(
        buf_lin, 32, guide, w0, buf_s0);
    // Stage 1: 64 -> 128, CI=4 CO=4, leaky
    pac_stage<4, 4, false><<<(4 * 128 * 128 + 255) / 256, 256, 0, stream>>>(
        buf_s0, 64, guide, w1, buf_s1);
    // Stage 2: 128 -> 256, CI=4 CO=4, leaky
    pac_stage<4, 4, false><<<(4 * 256 * 256 + 255) / 256, 256, 0, stream>>>(
        buf_s1, 128, guide, w2, buf_s2);
    // Stage 3: 256 -> 512, CI=4 CO=1, sigmoid -> output
    pac_stage<4, 1, true><<<(4 * 512 * 512 + 255) / 256, 256, 0, stream>>>(
        buf_s2, 256, guide, w3, outp);
}
